// Model_62852551410348
// MI455X (gfx1250) — matrix operand order repaired and checked by reading, not compiled, not run on hardware
//
#include <hip/hip_runtime.h>
#include <hip/hip_bf16.h>
#include <stdint.h>
#include <stddef.h>

// ---------------------------------------------------------------------------
// MI455X / gfx1250: 2-layer graph-conv GRU + HIPPO scan.
// All dense math via v_wmma_f32_16x16x32_bf16 (wave32 WMMA).
// All WMMA operands pre-swizzled into per-lane fragment order so loads are
// 32-byte vectors (global_load_b128 / ds_load_b128), not 2-byte gathers.
// ---------------------------------------------------------------------------

typedef __bf16 bf16_t;
typedef __attribute__((ext_vector_type(16))) __bf16 bf16x16;
typedef __attribute__((ext_vector_type(8)))  float  f32x8;

union CDu { f32x8 v; float e[8]; };

#define N_NODES 300
#define NPAD    320
#define BATCH   32
#define TSTEPS  96
#define HID     64
#define EMBD    10
#define ORD     8
#define HORZ    12

// Layer geometry
#define C0    65
#define KP0   160
#define LDZ0  2176      // 32*65=2080 padded to 17*128
#define GX0   17
#define C1    128
#define KP1   256
#define LDZ1  4096
#define GX1   32

#define KTM   (NPAD / 32)   // K-tiles over node dim (10)

// ---------------------------------------------------------------------------
// Swizzled-fragment helpers.
// A-tile (16Mx32K bf16): storage [tile][lane][16]; lane = h*16 + (row&15),
//   element i <-> ksub = 16*h + i.
// B-tile (32Kx16N bf16): storage [tile][lane][16]; lane = h*16 + (col&15),
//   element i <-> ksub = 16*h + i.
// ---------------------------------------------------------------------------
__device__ inline bf16x16 ld_frag(const bf16_t* tile_base) {
  int lane = threadIdx.x & 31;
  return *(const bf16x16*)(tile_base + (size_t)lane * 16);
}

__device__ inline f32x8 wmma_bf16(bf16x16 a, bf16x16 b, f32x8 c) {
  return __builtin_amdgcn_wmma_f32_16x16x32_bf16(false, a, false, b,
                                                 (short)0, c, false, false);
}

__device__ inline f32x8 f32x8_zero() {
  CDu z;
#pragma unroll
  for (int i = 0; i < 8; ++i) z.e[i] = 0.0f;
  return z.v;
}

// map (K-row ksub 0..31, col 0..15) -> element offset inside a 512-elem B tile
__device__ inline int btile_elem(int ksub, int col15) {
  int h = ksub >> 4;
  int i = ksub & 15;
  return (h * 16 + col15) * 16 + i;
}
// map (row 0..15, ksub 0..31) -> element offset inside a 512-elem A tile
__device__ inline int atile_elem(int row15, int ksub) {
  int h = ksub >> 4;
  int i = ksub & 15;
  return (h * 16 + row15) * 16 + i;
}

// ---------------------------------------------------------------------------
// Graph learning: adjacency from hard gumbel-softmax over edge MLP
// ---------------------------------------------------------------------------
__global__ void adj_kernel(const float* __restrict__ emb,
                           const float* __restrict__ W1, const float* __restrict__ b1,
                           const float* __restrict__ W2, const float* __restrict__ b2,
                           const float* __restrict__ gu, float* __restrict__ adj) {
  int p = blockIdx.x * blockDim.x + threadIdx.x;
  if (p >= N_NODES * N_NODES) return;
  int i = p / N_NODES, j = p % N_NODES;
  float hf[EMBD];
#pragma unroll
  for (int o = 0; o < EMBD; ++o) {
    float acc = b1[o];
#pragma unroll
    for (int e = 0; e < EMBD; ++e) acc += emb[j * EMBD + e] * W1[e * EMBD + o];
#pragma unroll
    for (int e = 0; e < EMBD; ++e) acc += emb[i * EMBD + e] * W1[(EMBD + e) * EMBD + o];
    hf[o] = fmaxf(acc, 0.0f);
  }
  float lg0 = b2[0], lg1 = b2[1];
#pragma unroll
  for (int o = 0; o < EMBD; ++o) { lg0 += hf[o] * W2[o * 2 + 0]; lg1 += hf[o] * W2[o * 2 + 1]; }
  float g0 = -logf(-logf(gu[p * 2 + 0] + 1e-20f) + 1e-20f);
  float g1 = -logf(-logf(gu[p * 2 + 1] + 1e-20f) + 1e-20f);
  float a0 = (lg0 + g0) / 1e-4f, a1 = (lg1 + g1) / 1e-4f;
  float mx = fmaxf(a0, a1);
  float e0 = __expf(a0 - mx), e1 = __expf(a1 - mx);
  float s0 = e0 / (e0 + e1);
  float hard0 = (a0 >= a1) ? 1.0f : 0.0f;
  float y0 = (hard0 - s0) + s0;
  adj[p] = (i == j) ? 0.0f : y0;
}

// ---------------------------------------------------------------------------
// M = 0.5*(softmax(relu(E E^T)) + adj/clip(rowsum,1,inf)) -> bf16 320x320
// ---------------------------------------------------------------------------
__device__ inline float dot_emb(const float* a, const float* b) {
  float s = 0.0f;
#pragma unroll
  for (int e = 0; e < EMBD; ++e) s += a[e] * b[e];
  return s;
}

__global__ __launch_bounds__(128) void build_m_kernel(const float* __restrict__ emb,
                                                      const float* __restrict__ adj,
                                                      bf16_t* __restrict__ M) {
  int n = blockIdx.x;
  int tid = threadIdx.x;
  if (n >= N_NODES) {
    for (int m = tid; m < NPAD; m += 128) M[(size_t)n * NPAD + m] = (bf16_t)0.0f;
    return;
  }
  __shared__ float red[128];
  const float* en = emb + n * EMBD;
  float lmax = -1e30f;
  for (int m = tid; m < N_NODES; m += 128)
    lmax = fmaxf(lmax, fmaxf(dot_emb(en, emb + m * EMBD), 0.0f));
  red[tid] = lmax; __syncthreads();
  for (int s = 64; s > 0; s >>= 1) { if (tid < s) red[tid] = fmaxf(red[tid], red[tid + s]); __syncthreads(); }
  float smax = red[0]; __syncthreads();
  float lsum = 0.0f;
  for (int m = tid; m < N_NODES; m += 128)
    lsum += __expf(fmaxf(dot_emb(en, emb + m * EMBD), 0.0f) - smax);
  red[tid] = lsum; __syncthreads();
  for (int s = 64; s > 0; s >>= 1) { if (tid < s) red[tid] += red[tid + s]; __syncthreads(); }
  float ssum = red[0]; __syncthreads();
  float lrs = 0.0f;
  for (int m = tid; m < N_NODES; m += 128) lrs += adj[n * N_NODES + m];
  red[tid] = lrs; __syncthreads();
  for (int s = 64; s > 0; s >>= 1) { if (tid < s) red[tid] += red[tid + s]; __syncthreads(); }
  float denom = fmaxf(red[0], 1.0f); __syncthreads();
  for (int m = tid; m < NPAD; m += 128) {
    float v = 0.0f;
    if (m < N_NODES) {
      float d = fmaxf(dot_emb(en, emb + m * EMBD), 0.0f);
      v = 0.5f * (__expf(d - smax) / ssum + adj[n * N_NODES + m] / denom);
    }
    M[(size_t)n * NPAD + m] = (bf16_t)v;
  }
}

// one-time A-tile swizzle of M: Msw[((rt*KTM+kt)*32+lane)*16+i]
__global__ void swizzle_m_kernel(const bf16_t* __restrict__ M, bf16_t* __restrict__ Msw) {
  int idx = blockIdx.x * blockDim.x + threadIdx.x;
  const int total = (NPAD / 16) * KTM * 512;
  if (idx >= total) return;
  int i = idx & 15;
  int lane = (idx >> 4) & 31;
  int tile = idx >> 9;
  int kt = tile % KTM;
  int rt = tile / KTM;
  int h = lane >> 4;
  int r = rt * 16 + (lane & 15);
  int ksub = (h << 4) | i;
  Msw[idx] = M[(size_t)r * NPAD + kt * 32 + ksub];
}

// ---------------------------------------------------------------------------
// Per-(b,n) mean / stdev over T
// ---------------------------------------------------------------------------
__global__ void meanstd_kernel(const float* __restrict__ x,
                               float* __restrict__ meanb, float* __restrict__ stdb) {
  int idx = blockIdx.x * blockDim.x + threadIdx.x;
  if (idx >= BATCH * N_NODES) return;
  int b = idx / N_NODES, n = idx % N_NODES;
  const float* xp = x + (size_t)b * TSTEPS * N_NODES + n;
  float s = 0.0f;
  for (int t = 0; t < TSTEPS; ++t) s += xp[(size_t)t * N_NODES];
  float mu = s / (float)TSTEPS;
  float v = 0.0f;
  for (int t = 0; t < TSTEPS; ++t) { float d = xp[(size_t)t * N_NODES] - mu; v += d * d; }
  v /= (float)TSTEPS;
  meanb[idx] = mu;
  stdb[idx]  = sqrtf(v + 1e-5f);
}

// ---------------------------------------------------------------------------
// Per-node weights, written directly in B-tile fragment order:
// dst flat: n*(KPAD*O) + ((kt*(O/16)+nt)*32 + lane)*16 + i
// logical j (K index) packs (support k=0 rows | k=1 rows | zero pad)
// ---------------------------------------------------------------------------
__global__ void mkw_kernel(const float* __restrict__ emb, const float* __restrict__ w,
                           bf16_t* __restrict__ dst, int CIN, int KPAD, int O) {
  size_t idx = (size_t)blockIdx.x * blockDim.x + threadIdx.x;
  size_t total = (size_t)N_NODES * KPAD * O;
  if (idx >= total) return;
  int per_node = KPAD * O;
  int n = (int)(idx / per_node);
  int r = (int)(idx % per_node);
  int i = r & 15;
  int lane = (r >> 4) & 31;
  int tile = r >> 9;
  int NT = O >> 4;
  int kt = tile / NT;
  int nt = tile % NT;
  int h = lane >> 4;
  int o = nt * 16 + (lane & 15);
  int j = kt * 32 + h * 16 + i;
  float acc = 0.0f;
  if (j < 2 * CIN) {
    int kk = (j < CIN) ? 0 : 1;
    int jj = (j < CIN) ? j : (j - CIN);
    const float* wp = w + ((size_t)kk * CIN + jj) * O + o;
    size_t estride = (size_t)2 * CIN * O;
#pragma unroll
    for (int e = 0; e < EMBD; ++e) acc += emb[n * EMBD + e] * wp[e * estride];
  }
  dst[idx] = (bf16_t)acc;
}

__global__ void mkb_kernel(const float* __restrict__ emb, const float* __restrict__ bp,
                           float* __restrict__ dst, int O) {
  int idx = blockIdx.x * blockDim.x + threadIdx.x;
  if (idx >= N_NODES * O) return;
  int n = idx / O, o = idx % O;
  float acc = 0.0f;
#pragma unroll
  for (int e = 0; e < EMBD; ++e) acc += emb[n * EMBD + e] * bp[e * O + o];
  dst[idx] = acc;
}

// ---------------------------------------------------------------------------
// Pack kernels: row-major ZP/CP (for gate LDS staging) + B-swizzled Zsw/Csw
// (for the graph-conv GEMM).  K-dim row = node m.
// ---------------------------------------------------------------------------
__device__ inline size_t bsw_off(int m, int col, int ldz) {
  int kt = m >> 5, ksub = m & 31;
  int nt = col >> 4;
  return ((size_t)(kt * (ldz >> 4) + nt)) * 512 + btile_elem(ksub, col & 15);
}

__global__ void pack0_kernel(const float* __restrict__ x, const float* __restrict__ meanb,
                             const float* __restrict__ stdb, const float* __restrict__ state,
                             bf16_t* __restrict__ ZP, bf16_t* __restrict__ CP,
                             bf16_t* __restrict__ Zsw, bf16_t* __restrict__ Csw,
                             int ldz, int t) {
  int idx = blockIdx.x * blockDim.x + threadIdx.x;
  int total = N_NODES * BATCH * C0;
  if (idx >= total) return;
  int m = idx / (BATCH * C0);
  int r = idx % (BATCH * C0);
  int b = r / C0, c = r % C0;
  float v;
  if (c == 0) {
    float mu = meanb[b * N_NODES + m], sd = stdb[b * N_NODES + m];
    const float* xp = x + (size_t)b * TSTEPS * N_NODES + m;
    float s0 = (xp[(size_t)t * N_NODES] - mu) / sd;
    if (t == 0 || t == TSTEPS - 1) v = s0;
    else {
      float sm = (xp[(size_t)(t - 1) * N_NODES] - mu) / sd;
      float sp = (xp[(size_t)(t + 1) * N_NODES] - mu) / sd;
      v = (sm + s0 + sp) / 3.0f;
    }
  } else {
    v = state[((size_t)b * N_NODES + m) * HID + (c - 1)];
  }
  bf16_t bv = (bf16_t)v;
  int col = b * C0 + c;
  size_t so = bsw_off(m, col, ldz);
  ZP[(size_t)m * ldz + col] = bv;
  Zsw[so] = bv;
  if (c == 0) { CP[(size_t)m * ldz + col] = bv; Csw[so] = bv; }
}

__global__ void pack1_kernel(const bf16_t* __restrict__ hs, const float* __restrict__ state,
                             bf16_t* __restrict__ ZP, bf16_t* __restrict__ CP,
                             bf16_t* __restrict__ Zsw, bf16_t* __restrict__ Csw,
                             int ldz, int t) {
  int idx = blockIdx.x * blockDim.x + threadIdx.x;
  int total = N_NODES * BATCH * C1;
  if (idx >= total) return;
  int m = idx / (BATCH * C1);
  int r = idx % (BATCH * C1);
  int b = r / C1, c = r % C1;
  float v;
  bool xpart = (c < HID);
  if (xpart) {
    size_t base = (((size_t)t * BATCH + b) * N_NODES + m) * HID + c;
    size_t stride_t = (size_t)BATCH * N_NODES * HID;
    float h0 = (float)hs[base];
    if (t == 0 || t == TSTEPS - 1) v = h0;
    else v = ((float)hs[base - stride_t] + h0 + (float)hs[base + stride_t]) / 3.0f;
  } else {
    v = state[((size_t)b * N_NODES + m) * HID + (c - HID)];
  }
  bf16_t bv = (bf16_t)v;
  int col = b * C1 + c;
  size_t so = bsw_off(m, col, ldz);
  ZP[(size_t)m * ldz + col] = bv;
  Zsw[so] = bv;
  if (xpart) { CP[(size_t)m * ldz + col] = bv; Csw[so] = bv; }
}

// ---------------------------------------------------------------------------
// Graph-conv GEMM: Y(320 x cols) = M(320x320) @ Z(320 x cols)
// A from Msw (A-frag order), B from Zsw (B-frag order), Y row-major bf16.
// 8 waves/block, 32x32 per wave, 64x128 per block.
// ---------------------------------------------------------------------------
__global__ __launch_bounds__(256) void conv_gemm_kernel(const bf16_t* __restrict__ Msw,
                                                        const bf16_t* __restrict__ Zsw,
                                                        bf16_t* __restrict__ Y, int ldz) {
  int wave = threadIdx.x >> 5;
  int wr = wave >> 2;
  int wc = wave & 3;
  int row0 = blockIdx.y * 64 + wr * 32;
  int col0 = blockIdx.x * 128 + wc * 32;
  int rt = row0 >> 4;
  int nt = col0 >> 4;
  int NT = ldz >> 4;
  f32x8 acc00 = f32x8_zero(), acc01 = f32x8_zero();
  f32x8 acc10 = f32x8_zero(), acc11 = f32x8_zero();
  for (int kt = 0; kt < KTM; ++kt) {
    bf16x16 a0 = ld_frag(Msw + ((size_t)(rt * KTM + kt)) * 512);
    bf16x16 a1 = ld_frag(Msw + ((size_t)((rt + 1) * KTM + kt)) * 512);
    bf16x16 b0 = ld_frag(Zsw + ((size_t)(kt * NT + nt)) * 512);
    bf16x16 b1 = ld_frag(Zsw + ((size_t)(kt * NT + nt + 1)) * 512);
    acc00 = wmma_bf16(a0, b0, acc00);
    acc01 = wmma_bf16(a0, b1, acc01);
    acc10 = wmma_bf16(a1, b0, acc10);
    acc11 = wmma_bf16(a1, b1, acc11);
  }
  int lane = threadIdx.x & 31;
  int cn = lane & 15, hh = lane >> 4;
  const f32x8 accs[4] = {acc00, acc01, acc10, acc11};
#pragma unroll
  for (int i = 0; i < 2; ++i)
#pragma unroll
    for (int j = 0; j < 2; ++j) {
      CDu cd; cd.v = accs[i * 2 + j];
#pragma unroll
      for (int vr = 0; vr < 8; ++vr) {
        int rr = row0 + i * 16 + vr + 8 * hh;
        int cc = col0 + j * 16 + cn;
        Y[(size_t)rr * ldz + cc] = (bf16_t)cd.e[vr];
      }
    }
}

// ---------------------------------------------------------------------------
// Gate z/r per node: zr = sigmoid(Zcat @ Wg[n] + bg[n])
//   z (o<64)  -> zbuf f32
//   r (o>=64) -> cin = r*state -> CP (row-major) + Csw (B-frag order)
// LDS Zcat stored in A-frag order -> single 32B ds vector loads.
// ---------------------------------------------------------------------------
template <int C, int KPAD>
__global__ __launch_bounds__(256) void gate_zr_kernel(const bf16_t* __restrict__ ZP,
                                                      const bf16_t* __restrict__ YZ, int ldz,
                                                      const bf16_t* __restrict__ W,
                                                      const float* __restrict__ bg,
                                                      const float* __restrict__ state,
                                                      float* __restrict__ zbuf,
                                                      bf16_t* __restrict__ CP,
                                                      bf16_t* __restrict__ Csw) {
  constexpr int KT = KPAD / 32;
  __shared__ __align__(32) bf16_t zc[BATCH * KPAD];
  int n = blockIdx.x;
  for (int idx = threadIdx.x; idx < BATCH * KPAD; idx += 256) {
    int b = idx / KPAD, j = idx % KPAD;
    bf16_t v = (bf16_t)0.0f;
    if (j < C)          v = ZP[(size_t)n * ldz + b * C + j];
    else if (j < 2 * C) v = YZ[(size_t)n * ldz + b * C + (j - C)];
    int mt = b >> 4, kt = j >> 5;
    zc[(size_t)(mt * KT + kt) * 512 + atile_elem(b & 15, j & 31)] = v;
  }
  __syncthreads();
  int wave = threadIdx.x >> 5;
  int mt = wave & 1;
  int ct0 = (wave >> 1) * 2;
  f32x8 c0 = f32x8_zero(), c1 = f32x8_zero();
  const bf16_t* Wn = W + (size_t)n * KPAD * 128;
  for (int kt = 0; kt < KT; ++kt) {
    bf16x16 a  = ld_frag(zc + (size_t)(mt * KT + kt) * 512);
    bf16x16 b0 = ld_frag(Wn + (size_t)(kt * 8 + ct0) * 512);
    bf16x16 b1 = ld_frag(Wn + (size_t)(kt * 8 + ct0 + 1) * 512);
    c0 = wmma_bf16(a, b0, c0);
    c1 = wmma_bf16(a, b1, c1);
  }
  int lane = threadIdx.x & 31;
  int cn = lane & 15, hh = lane >> 4;
#pragma unroll
  for (int j = 0; j < 2; ++j) {
    int o = (ct0 + j) * 16 + cn;
    float bias = bg[n * 128 + o];
    CDu cd; cd.v = j ? c1 : c0;
#pragma unroll
    for (int vr = 0; vr < 8; ++vr) {
      int b = mt * 16 + vr + 8 * hh;
      float s = 1.0f / (1.0f + __expf(-(cd.e[vr] + bias)));
      if (o < HID) {
        zbuf[((size_t)b * N_NODES + n) * HID + o] = s;
      } else {
        int oo = o - HID;
        float rv = s * state[((size_t)b * N_NODES + n) * HID + oo];
        int cidx = (C == C0) ? (1 + oo) : (HID + oo);
        int col = b * C + cidx;
        bf16_t bv = (bf16_t)rv;
        CP[(size_t)n * ldz + col] = bv;
        Csw[bsw_off(n, col, ldz)] = bv;
      }
    }
  }
}

// ---------------------------------------------------------------------------
// Gate hc per node: hc = tanh(Ccat @ Wc[n] + bc[n] + c@wh)
// h = z*state + (1-z)*hc ; writes new state (+ hs sequence for layer 0)
// ---------------------------------------------------------------------------
template <int C, int KPAD>
__global__ __launch_bounds__(256) void gate_hc_kernel(const bf16_t* __restrict__ CP,
                                                      const bf16_t* __restrict__ YC, int ldz,
                                                      const bf16_t* __restrict__ W,
                                                      const float* __restrict__ bc,
                                                      const float* __restrict__ wh,
                                                      const float* __restrict__ zbuf,
                                                      const float* __restrict__ chip,
                                                      float* __restrict__ state,
                                                      bf16_t* __restrict__ hs, int t) {
  constexpr int KT = KPAD / 32;
  __shared__ __align__(32) bf16_t zc[BATCH * KPAD];
  int n = blockIdx.x;
  for (int idx = threadIdx.x; idx < BATCH * KPAD; idx += 256) {
    int b = idx / KPAD, j = idx % KPAD;
    bf16_t v = (bf16_t)0.0f;
    if (j < C)          v = CP[(size_t)n * ldz + b * C + j];
    else if (j < 2 * C) v = YC[(size_t)n * ldz + b * C + (j - C)];
    int mt = b >> 4, kt = j >> 5;
    zc[(size_t)(mt * KT + kt) * 512 + atile_elem(b & 15, j & 31)] = v;
  }
  __syncthreads();
  int wave = threadIdx.x >> 5;
  int mt = wave & 1;
  int ct = wave >> 1;
  f32x8 acc = f32x8_zero();
  const bf16_t* Wn = W + (size_t)n * KPAD * HID;
  for (int kt = 0; kt < KT; ++kt) {
    bf16x16 a = ld_frag(zc + (size_t)(mt * KT + kt) * 512);
    bf16x16 b = ld_frag(Wn + (size_t)(kt * 4 + ct) * 512);
    acc = wmma_bf16(a, b, acc);
  }
  int lane = threadIdx.x & 31;
  int cn = lane & 15, hh = lane >> 4;
  int o = ct * 16 + cn;
  float bias = bc[n * HID + o];
  CDu cd; cd.v = acc;
#pragma unroll
  for (int vr = 0; vr < 8; ++vr) {
    int b = mt * 16 + vr + 8 * hh;
    size_t sidx = ((size_t)b * N_NODES + n) * HID + o;
    const float* cb = chip + ((size_t)b * N_NODES + n) * ORD;
    float cw = 0.0f;
#pragma unroll
    for (int j = 0; j < ORD; ++j) cw += cb[j] * wh[j * HID + o];
    float hcv = tanhf(cd.e[vr] + bias + cw);
    float z = zbuf[sidx];
    float st = state[sidx];
    float hv = z * st + (1.0f - z) * hcv;
    state[sidx] = hv;
    if (hs) hs[(((size_t)t * BATCH + b) * N_NODES + n) * HID + o] = (bf16_t)hv;
  }
}

// ---------------------------------------------------------------------------
// HIPPO side-state: c = c - (c @ A^T)/tt + mean(h)*Bvec/tt
// ---------------------------------------------------------------------------
__global__ void hippo_kernel(const float* __restrict__ state, float* __restrict__ chip,
                             float tt) {
  int idx = blockIdx.x * blockDim.x + threadIdx.x;
  if (idx >= BATCH * N_NODES) return;
  const float* h = state + (size_t)idx * HID;
  float m = 0.0f;
  for (int o = 0; o < HID; ++o) m += h[o];
  m /= (float)HID;
  float c[ORD], rr[ORD];
#pragma unroll
  for (int i = 0; i < ORD; ++i) { c[i] = chip[(size_t)idx * ORD + i]; rr[i] = sqrtf(2.0f * i + 1.0f); }
  float cn[ORD];
#pragma unroll
  for (int i = 0; i < ORD; ++i) {
    float acc = 0.0f;
#pragma unroll
    for (int j = 0; j < ORD; ++j) {
      float A = (j < i) ? rr[i] * rr[j] : ((i == j) ? (float)(i + 1) : 0.0f);
      acc += A * c[j];
    }
    cn[i] = c[i] - acc / tt + m * rr[i] / tt;
  }
#pragma unroll
  for (int i = 0; i < ORD; ++i) chip[(size_t)idx * ORD + i] = cn[i];
}

// ---------------------------------------------------------------------------
// Output head
// ---------------------------------------------------------------------------
__global__ void out_kernel(const float* __restrict__ state,
                           const float* __restrict__ end_w, const float* __restrict__ end_b,
                           const float* __restrict__ meanb, const float* __restrict__ stdb,
                           float* __restrict__ out) {
  int idx = blockIdx.x * blockDim.x + threadIdx.x;
  if (idx >= BATCH * HORZ * N_NODES) return;
  int b = idx / (HORZ * N_NODES);
  int p = (idx / N_NODES) % HORZ;
  int n = idx % N_NODES;
  const float* h = state + ((size_t)b * N_NODES + n) * HID;
  float acc = end_b[p];
#pragma unroll 8
  for (int o = 0; o < HID; ++o) acc += h[o] * end_w[p * HID + o];
  out[idx] = acc * stdb[b * N_NODES + n] + meanb[b * N_NODES + n];
}

// ---------------------------------------------------------------------------
// Host launch
// ---------------------------------------------------------------------------
static inline size_t align256(size_t x) { return (x + 255) & ~(size_t)255; }

extern "C" void kernel_launch(void* const* d_in, const int* in_sizes, int n_in,
                              void* d_out, int out_size, void* d_ws, size_t ws_size,
                              hipStream_t stream) {
  (void)in_sizes; (void)n_in; (void)out_size; (void)ws_size;

  const float* x_enc    = (const float*)d_in[0];
  const float* node_emb = (const float*)d_in[4];
  const float* fc_out_w = (const float*)d_in[5];
  const float* fc_out_b = (const float*)d_in[6];
  const float* fc_cat_w = (const float*)d_in[7];
  const float* fc_cat_b = (const float*)d_in[8];
  const float* gumbel_u = (const float*)d_in[9];
  const float* wg0 = (const float*)d_in[10];
  const float* bg0p = (const float*)d_in[11];
  const float* wc0 = (const float*)d_in[12];
  const float* bc0p = (const float*)d_in[13];
  const float* wh0 = (const float*)d_in[14];
  const float* wg1 = (const float*)d_in[15];
  const float* bg1p = (const float*)d_in[16];
  const float* wc1 = (const float*)d_in[17];
  const float* bc1p = (const float*)d_in[18];
  const float* wh1 = (const float*)d_in[19];
  const float* end_w = (const float*)d_in[20];
  const float* end_b = (const float*)d_in[21];

  char* wsb = (char*)d_ws;
  size_t off = 0;
  auto alloc = [&](size_t bytes) -> void* { void* p = wsb + off; off = align256(off + bytes); return p; };

  float*  adj   = (float*)alloc((size_t)N_NODES * N_NODES * 4);
  bf16_t* Mb    = (bf16_t*)alloc((size_t)NPAD * NPAD * 2);
  bf16_t* Msw   = (bf16_t*)alloc((size_t)NPAD * NPAD * 2);
  float*  meanb = (float*)alloc((size_t)BATCH * N_NODES * 4);
  float*  stdb  = (float*)alloc((size_t)BATCH * N_NODES * 4);
  float*  state = (float*)alloc((size_t)BATCH * N_NODES * HID * 4);
  float*  chip  = (float*)alloc((size_t)BATCH * N_NODES * ORD * 4);
  float*  zbuf  = (float*)alloc((size_t)BATCH * N_NODES * HID * 4);
  bf16_t* ZP    = (bf16_t*)alloc((size_t)NPAD * LDZ1 * 2);
  bf16_t* CPb   = (bf16_t*)alloc((size_t)NPAD * LDZ1 * 2);
  bf16_t* Zsw   = (bf16_t*)alloc((size_t)NPAD * LDZ1 * 2);
  bf16_t* Csw   = (bf16_t*)alloc((size_t)NPAD * LDZ1 * 2);
  bf16_t* YZ    = (bf16_t*)alloc((size_t)NPAD * LDZ1 * 2);
  bf16_t* YC    = (bf16_t*)alloc((size_t)NPAD * LDZ1 * 2);
  bf16_t* Wg0b  = (bf16_t*)alloc((size_t)N_NODES * KP0 * 128 * 2);
  bf16_t* Wc0b  = (bf16_t*)alloc((size_t)N_NODES * KP0 * HID * 2);
  bf16_t* Wg1b  = (bf16_t*)alloc((size_t)N_NODES * KP1 * 128 * 2);
  bf16_t* Wc1b  = (bf16_t*)alloc((size_t)N_NODES * KP1 * HID * 2);
  float*  bg0n  = (float*)alloc((size_t)N_NODES * 128 * 4);
  float*  bc0n  = (float*)alloc((size_t)N_NODES * HID * 4);
  float*  bg1n  = (float*)alloc((size_t)N_NODES * 128 * 4);
  float*  bc1n  = (float*)alloc((size_t)N_NODES * HID * 4);
  bf16_t* hs    = (bf16_t*)alloc((size_t)TSTEPS * BATCH * N_NODES * HID * 2);

  hipMemsetAsync(state, 0, (size_t)BATCH * N_NODES * HID * 4, stream);
  hipMemsetAsync(chip,  0, (size_t)BATCH * N_NODES * ORD * 4, stream);
  hipMemsetAsync(ZP,  0, (size_t)NPAD * LDZ1 * 2, stream);
  hipMemsetAsync(CPb, 0, (size_t)NPAD * LDZ1 * 2, stream);
  hipMemsetAsync(Zsw, 0, (size_t)NPAD * LDZ1 * 2, stream);
  hipMemsetAsync(Csw, 0, (size_t)NPAD * LDZ1 * 2, stream);
  hipMemsetAsync(YZ,  0, (size_t)NPAD * LDZ1 * 2, stream);
  hipMemsetAsync(YC,  0, (size_t)NPAD * LDZ1 * 2, stream);

  adj_kernel<<<(N_NODES * N_NODES + 255) / 256, 256, 0, stream>>>(
      node_emb, fc_out_w, fc_out_b, fc_cat_w, fc_cat_b, gumbel_u, adj);
  build_m_kernel<<<NPAD, 128, 0, stream>>>(node_emb, adj, Mb);
  swizzle_m_kernel<<<((NPAD / 16) * KTM * 512 + 255) / 256, 256, 0, stream>>>(Mb, Msw);
  meanstd_kernel<<<(BATCH * N_NODES + 255) / 256, 256, 0, stream>>>(x_enc, meanb, stdb);

  {
    size_t t;
    t = (size_t)N_NODES * KP0 * 128;
    mkw_kernel<<<(unsigned)((t + 255) / 256), 256, 0, stream>>>(node_emb, wg0, Wg0b, C0, KP0, 128);
    t = (size_t)N_NODES * KP0 * HID;
    mkw_kernel<<<(unsigned)((t + 255) / 256), 256, 0, stream>>>(node_emb, wc0, Wc0b, C0, KP0, HID);
    t = (size_t)N_NODES * KP1 * 128;
    mkw_kernel<<<(unsigned)((t + 255) / 256), 256, 0, stream>>>(node_emb, wg1, Wg1b, C1, KP1, 128);
    t = (size_t)N_NODES * KP1 * HID;
    mkw_kernel<<<(unsigned)((t + 255) / 256), 256, 0, stream>>>(node_emb, wc1, Wc1b, C1, KP1, HID);
  }
  mkb_kernel<<<(N_NODES * 128 + 255) / 256, 256, 0, stream>>>(node_emb, bg0p, bg0n, 128);
  mkb_kernel<<<(N_NODES * HID + 255) / 256, 256, 0, stream>>>(node_emb, bc0p, bc0n, HID);
  mkb_kernel<<<(N_NODES * 128 + 255) / 256, 256, 0, stream>>>(node_emb, bg1p, bg1n, 128);
  mkb_kernel<<<(N_NODES * HID + 255) / 256, 256, 0, stream>>>(node_emb, bc1p, bc1n, HID);

  // ---- layer 0 scan ----
  for (int t = 0; t < TSTEPS; ++t) {
    pack0_kernel<<<(N_NODES * BATCH * C0 + 255) / 256, 256, 0, stream>>>(
        x_enc, meanb, stdb, state, ZP, CPb, Zsw, Csw, LDZ0, t);
    conv_gemm_kernel<<<dim3(GX0, NPAD / 64), 256, 0, stream>>>(Msw, Zsw, YZ, LDZ0);
    gate_zr_kernel<C0, KP0><<<N_NODES, 256, 0, stream>>>(
        ZP, YZ, LDZ0, Wg0b, bg0n, state, zbuf, CPb, Csw);
    conv_gemm_kernel<<<dim3(GX0, NPAD / 64), 256, 0, stream>>>(Msw, Csw, YC, LDZ0);
    gate_hc_kernel<C0, KP0><<<N_NODES, 256, 0, stream>>>(
        CPb, YC, LDZ0, Wc0b, bc0n, wh0, zbuf, chip, state, hs, t);
    hippo_kernel<<<(BATCH * N_NODES + 255) / 256, 256, 0, stream>>>(state, chip, (float)(t + 1));
  }

  hipMemsetAsync(state, 0, (size_t)BATCH * N_NODES * HID * 4, stream);
  hipMemsetAsync(chip,  0, (size_t)BATCH * N_NODES * ORD * 4, stream);

  // ---- layer 1 scan ----
  for (int t = 0; t < TSTEPS; ++t) {
    pack1_kernel<<<(N_NODES * BATCH * C1 + 255) / 256, 256, 0, stream>>>(
        hs, state, ZP, CPb, Zsw, Csw, LDZ1, t);
    conv_gemm_kernel<<<dim3(GX1, NPAD / 64), 256, 0, stream>>>(Msw, Zsw, YZ, LDZ1);
    gate_zr_kernel<C1, KP1><<<N_NODES, 256, 0, stream>>>(
        ZP, YZ, LDZ1, Wg1b, bg1n, state, zbuf, CPb, Csw);
    conv_gemm_kernel<<<dim3(GX1, NPAD / 64), 256, 0, stream>>>(Msw, Csw, YC, LDZ1);
    gate_hc_kernel<C1, KP1><<<N_NODES, 256, 0, stream>>>(
        CPb, YC, LDZ1, Wc1b, bc1n, wh1, zbuf, chip, state, (bf16_t*)nullptr, t);
    hippo_kernel<<<(BATCH * N_NODES + 255) / 256, 256, 0, stream>>>(state, chip, (float)(t + 1));
  }

  out_kernel<<<(BATCH * HORZ * N_NODES + 255) / 256, 256, 0, stream>>>(
      state, end_w, end_b, meanb, stdb, (float*)d_out);
}
